// GNNTracker_43825846288528
// MI455X (gfx1250) — matrix operand order repaired and checked by reading, not compiled, not run on hardware
//
#include <hip/hip_runtime.h>
#include <hip/hip_bf16.h>

// ---------------------------------------------------------------------------
// CDNA5 (gfx1250, wave32) GNN edge-classifier pipeline.
// Dense math goes through v_wmma_f32_16x16x32_bf16 (f32 accumulate).
// ---------------------------------------------------------------------------

typedef __attribute__((ext_vector_type(16))) __bf16 v16bf;
typedef __attribute__((ext_vector_type(16))) float  v16f;
typedef __attribute__((ext_vector_type(8)))  float  v8f;

#define HDIM 128

// Packed f32x16 -> bf16x16 (compiler lowers fptrunc to native packed cvt)
static __device__ __forceinline__ v16bf cvt16(float4 c0, float4 c1, float4 c2, float4 c3) {
  v16f t;
  t[0]=c0.x;  t[1]=c0.y;  t[2]=c0.z;  t[3]=c0.w;
  t[4]=c1.x;  t[5]=c1.y;  t[6]=c1.z;  t[7]=c1.w;
  t[8]=c2.x;  t[9]=c2.y;  t[10]=c2.z; t[11]=c2.w;
  t[12]=c3.x; t[13]=c3.y; t[14]=c3.z; t[15]=c3.w;
  return __builtin_convertvector(t, v16bf);
}

// ---------------------------------------------------------------------------
// small helper kernels
// ---------------------------------------------------------------------------
__global__ void k_cvt_bf16(const float* __restrict__ s, __bf16* __restrict__ d, int n) {
  int i = blockIdx.x * blockDim.x + threadIdx.x;
  if (i < n) d[i] = (__bf16)s[i];
}

__global__ void k_fill1(float* __restrict__ d, int n) {
  int i = blockIdx.x * blockDim.x + threadIdx.x;
  if (i < n) d[i] = 1.0f;   // self-loop contribution to degree
}

__global__ void k_deg_accum(const long long* __restrict__ dsti, float* __restrict__ deg, int e) {
  int i = blockIdx.x * blockDim.x + threadIdx.x;
  if (i < e) atomicAdd(&deg[(int)dsti[i]], 1.0f);
}

__global__ void k_rsqrt(float* __restrict__ d, int n) {
  int i = blockIdx.x * blockDim.x + threadIdx.x;
  if (i < n) d[i] = rsqrtf(d[i]);   // deg >= 1 always (self-loops)
}

__global__ void k_init_bias(float* __restrict__ o, const float* __restrict__ b, long long total) {
  long long i = (long long)blockIdx.x * blockDim.x + threadIdx.x;
  if (i < total) o[i] = b[(int)(i & (HDIM - 1))];
}

__global__ void k_relu(float* __restrict__ d, long long n) {
  long long i = (long long)blockIdx.x * blockDim.x + threadIdx.x;
  if (i < n) d[i] = fmaxf(d[i], 0.0f);
}

// one wave per (edge or self-loop): out[dst] += norm * xw[src]  (128 floats)
// x matrix (25.6 MB) is L2-resident on MI455X (192 MB L2): gathers + atomics hit L2.
__global__ __launch_bounds__(256)
void k_aggregate(const float* __restrict__ xw,
                 const long long* __restrict__ src, const long long* __restrict__ dst,
                 const float* __restrict__ dis, float* __restrict__ out, int E, int N) {
  long long t = (long long)blockIdx.x * 256 + threadIdx.x;
  int gw = (int)(t >> 5);
  int lane = threadIdx.x & 31;
  if (gw >= E + N) return;
  int s, d;
  if (gw < E) { s = (int)src[gw]; d = (int)dst[gw]; }
  else        { s = d = gw - E; }
  float nrm = dis[s] * dis[d];
  float4 v = *(const float4*)(xw + (size_t)s * HDIM + lane * 4);
  float* o = out + (size_t)d * HDIM + lane * 4;
  atomicAdd(o + 0, v.x * nrm);
  atomicAdd(o + 1, v.y * nrm);
  atomicAdd(o + 2, v.z * nrm);
  atomicAdd(o + 3, v.w * nrm);
}

// ---------------------------------------------------------------------------
// WMMA GEMM: Y(M x 128) = act(A(M x K) @ W(K x 128) + bias)
// block = 256 threads = 8 waves; each wave: RS=2 strips of 16 rows (32 rows),
// so each LDS B-fragment read is reused RS times. Block covers 256 rows.
// W panel (bf16) staged in LDS; A converted f32->bf16 with packed cvt.
// Fragment layouts per ISA 7.12.2 (wave32):
//   A 16x32 bf16: lane -> M=lane&15; lane-half selects K-chunks {0..7,16..23}/{8..15,24..31}
//   B 32x16 bf16: lane -> N=lane&15; lane-half selects K-chunks {0..7,16..23}/{8..15,24..31}
//   C/D 16x16 f32: vgpr i, lane -> [M = i + 8*(lane>>4), N = lane&15]
// ---------------------------------------------------------------------------
template<int KT, int ACT>
__global__ __launch_bounds__(256)
void gemm_bf16_nh(const float* __restrict__ A, const __bf16* __restrict__ W,
                  const float* __restrict__ bias, float* __restrict__ Y, int M) {
  constexpr int K  = KT * 32;
  constexpr int RS = 2;                       // row-strips per wave
  __shared__ alignas(32) __bf16 sW[K * HDIM];
  {
    const uint4* s = (const uint4*)W;
    uint4* d = (uint4*)sW;
    const int n16 = K * HDIM * 2 / 16;
    for (int i = threadIdx.x; i < n16; i += 256) d[i] = s[i];
  }
  __syncthreads();

  const int lane   = threadIdx.x & 31;
  const int wave   = threadIdx.x >> 5;
  const int lanelo = lane & 15;
  const int lanehi = lane >> 4;
  const int row0   = blockIdx.x * (128 * RS) + wave * (16 * RS);

  v16bf afrag[RS][KT];
#pragma unroll
  for (int r = 0; r < RS; ++r) {
    const int m = row0 + r * 16 + lanelo;
    const bool valid = (m < M);
    const float* arow = A + (size_t)m * K;
#pragma unroll
    for (int kt = 0; kt < KT; ++kt) {
      const int kb = kt * 32 + (lanehi << 3);
      float4 c0 = make_float4(0.f, 0.f, 0.f, 0.f), c1 = c0, c2 = c0, c3 = c0;
      if (valid) {
        c0 = *(const float4*)(arow + kb);
        c1 = *(const float4*)(arow + kb + 4);
        c2 = *(const float4*)(arow + kb + 16);
        c3 = *(const float4*)(arow + kb + 20);
      }
      afrag[r][kt] = cvt16(c0, c1, c2, c3);
    }
  }

#pragma unroll
  for (int nt = 0; nt < 8; ++nt) {
    v16bf bfrag[KT];
#pragma unroll
    for (int kt = 0; kt < KT; ++kt)
#pragma unroll
      for (int s = 0; s < 16; ++s)
        bfrag[kt][s] = sW[(kt * 32 + (lanehi << 3) + (s & 7) + ((s >> 3) << 4)) * HDIM + nt * 16 + lanelo];
    const float bv = bias ? bias[nt * 16 + lanelo] : 0.0f;
#pragma unroll
    for (int r = 0; r < RS; ++r) {
      v8f acc = {};
#pragma unroll
      for (int kt = 0; kt < KT; ++kt)
        acc = __builtin_amdgcn_wmma_f32_16x16x32_bf16(false, afrag[r][kt], false, bfrag[kt],
                                                      (short)0, acc, false, false);
#pragma unroll
      for (int i = 0; i < 8; ++i) {
        const int mm = row0 + r * 16 + i + (lanehi << 3);
        if (mm < M) {
          float v = acc[i] + bv;
          if (ACT) v = fmaxf(v, 0.0f);
          Y[(size_t)mm * HDIM + nt * 16 + lanelo] = v;
        }
      }
    }
  }
}

// ---------------------------------------------------------------------------
// Fused edge classifier: for each edge e,
//   h = relu([x[src]‖x[dst]] @ W1 + b1);  out[e] = sigmoid(h·w2 + b2)
// Gather fused into WMMA A-fragment load. RS=2 strips -> 32 edges per wave.
// ---------------------------------------------------------------------------
__global__ __launch_bounds__(256)
void edge_cls(const float* __restrict__ x,
              const long long* __restrict__ src, const long long* __restrict__ dst,
              const __bf16* __restrict__ W1, const float* __restrict__ b1,
              const float* __restrict__ w2, const float* __restrict__ b2,
              float* __restrict__ out, int E) {
  constexpr int RS = 2;
  __shared__ alignas(32) __bf16 sW[256 * HDIM];
  {
    const uint4* s = (const uint4*)W1;
    uint4* d = (uint4*)sW;
    const int n16 = 256 * HDIM * 2 / 16;
    for (int i = threadIdx.x; i < n16; i += 256) d[i] = s[i];
  }
  __syncthreads();

  const int lane   = threadIdx.x & 31;
  const int wave   = threadIdx.x >> 5;
  const int lanelo = lane & 15;
  const int lanehi = lane >> 4;
  const int e0     = blockIdx.x * (128 * RS) + wave * (16 * RS);

  v16bf afrag[RS][8];
#pragma unroll
  for (int r = 0; r < RS; ++r) {
    const int m = e0 + r * 16 + lanelo;
    const bool valid = (m < E);
    const int si = valid ? (int)src[m] : 0;
    const int di = valid ? (int)dst[m] : 0;
    const float* xs = x + (size_t)si * HDIM;
    const float* xd = x + (size_t)di * HDIM;
#pragma unroll
    for (int kt = 0; kt < 8; ++kt) {
      const int kb = kt * 32 + (lanehi << 3);
      const float* base = (kb < HDIM) ? xs : xd;   // concat: K<128 -> src row, else dst row
      const int off = kb & (HDIM - 1);
      float4 c0 = make_float4(0.f, 0.f, 0.f, 0.f), c1 = c0, c2 = c0, c3 = c0;
      if (valid) {
        c0 = *(const float4*)(base + off);
        c1 = *(const float4*)(base + off + 4);
        c2 = *(const float4*)(base + off + 16);
        c3 = *(const float4*)(base + off + 20);
      }
      afrag[r][kt] = cvt16(c0, c1, c2, c3);
    }
  }

  float partial[RS][8];
#pragma unroll
  for (int r = 0; r < RS; ++r)
#pragma unroll
    for (int i = 0; i < 8; ++i) partial[r][i] = 0.0f;

#pragma unroll
  for (int nt = 0; nt < 8; ++nt) {
    v16bf bfrag[8];
#pragma unroll
    for (int kt = 0; kt < 8; ++kt)
#pragma unroll
      for (int s = 0; s < 16; ++s)
        bfrag[kt][s] = sW[(kt * 32 + (lanehi << 3) + (s & 7) + ((s >> 3) << 4)) * HDIM + nt * 16 + lanelo];
    const float bv = b1[nt * 16 + lanelo];
    const float wv = w2[nt * 16 + lanelo];
#pragma unroll
    for (int r = 0; r < RS; ++r) {
      v8f acc = {};
#pragma unroll
      for (int kt = 0; kt < 8; ++kt)
        acc = __builtin_amdgcn_wmma_f32_16x16x32_bf16(false, afrag[r][kt], false, bfrag[kt],
                                                      (short)0, acc, false, false);
#pragma unroll
      for (int i = 0; i < 8; ++i) {
        float h = fmaxf(acc[i] + bv, 0.0f);         // layer-1 ReLU
        partial[r][i] = fmaf(h, wv, partial[r][i]); // layer-2 dot (per-lane slice)
      }
    }
  }
  // reduce across the 16 lanes that share the same output rows
#pragma unroll
  for (int ofs = 1; ofs < 16; ofs <<= 1) {
#pragma unroll
    for (int r = 0; r < RS; ++r)
#pragma unroll
      for (int i = 0; i < 8; ++i) partial[r][i] += __shfl_xor(partial[r][i], ofs, 32);
  }
  if (lanelo == 0) {
    const float bb = b2[0];
#pragma unroll
    for (int r = 0; r < RS; ++r)
#pragma unroll
      for (int i = 0; i < 8; ++i) {
        const int mm = e0 + r * 16 + i + (lanehi << 3);
        if (mm < E) out[mm] = 1.0f / (1.0f + __expf(-(partial[r][i] + bb)));
      }
  }
}

// ---------------------------------------------------------------------------
extern "C" void kernel_launch(void* const* d_in, const int* in_sizes, int n_in,
                              void* d_out, int out_size, void* d_ws, size_t ws_size,
                              hipStream_t stream) {
  (void)n_in; (void)out_size; (void)ws_size;
  const float*     nf     = (const float*)d_in[0];
  const long long* ei     = (const long long*)d_in[1];
  const float* enc_w1 = (const float*)d_in[2];
  const float* enc_b1 = (const float*)d_in[3];
  const float* enc_w2 = (const float*)d_in[4];
  const float* enc_b2 = (const float*)d_in[5];
  const float* conv_w[3] = {(const float*)d_in[6], (const float*)d_in[8], (const float*)d_in[10]};
  const float* conv_b[3] = {(const float*)d_in[7], (const float*)d_in[9], (const float*)d_in[11]};
  const float* cls_w1 = (const float*)d_in[12];
  const float* cls_b1 = (const float*)d_in[13];
  const float* cls_w2 = (const float*)d_in[14];
  const float* cls_b2 = (const float*)d_in[15];

  const int FD = 256;
  const int N = in_sizes[0] / FD;
  const int E = in_sizes[1] / 2;
  const long long* srci = ei;
  const long long* dsti = ei + E;
  float* out = (float*)d_out;

  // workspace carve
  char* p = (char*)d_ws;
  auto carve = [&](size_t bytes) -> char* {
    char* r = p; p += (bytes + 255) & ~(size_t)255; return r;
  };
  __bf16* wb_e1  = (__bf16*)carve((size_t)256 * 128 * 2);
  __bf16* wb_e2  = (__bf16*)carve((size_t)128 * 128 * 2);
  __bf16* wb_c0  = (__bf16*)carve((size_t)128 * 128 * 2);
  __bf16* wb_c1  = (__bf16*)carve((size_t)128 * 128 * 2);
  __bf16* wb_c2  = (__bf16*)carve((size_t)128 * 128 * 2);
  __bf16* wb_cls = (__bf16*)carve((size_t)256 * 128 * 2);
  __bf16* wb_c[3] = {wb_c0, wb_c1, wb_c2};
  float* dis = (float*)carve((size_t)N * 4);
  float* t0  = (float*)carve((size_t)N * HDIM * 4);   // xw scratch
  float* t1  = (float*)carve((size_t)N * HDIM * 4);   // x ping
  float* t2  = (float*)carve((size_t)N * HDIM * 4);   // x pong

  // weight conversion f32 -> bf16
  auto cvt = [&](const float* s, __bf16* d, int n) {
    k_cvt_bf16<<<(n + 255) / 256, 256, 0, stream>>>(s, d, n);
  };
  cvt(enc_w1, wb_e1, 256 * 128);
  cvt(enc_w2, wb_e2, 128 * 128);
  for (int l = 0; l < 3; ++l) cvt(conv_w[l], wb_c[l], 128 * 128);
  cvt(cls_w1, wb_cls, 256 * 128);

  // symmetric normalization: dis = rsqrt(in_deg(dst) + 1)
  k_fill1    <<<(N + 255) / 256, 256, 0, stream>>>(dis, N);
  k_deg_accum<<<(E + 255) / 256, 256, 0, stream>>>(dsti, dis, E);
  k_rsqrt    <<<(N + 255) / 256, 256, 0, stream>>>(dis, N);

  const int gN = (N + 255) / 256;   // 256 rows per block (RS=2)
  // node encoder
  gemm_bf16_nh<8, 1><<<gN, 256, 0, stream>>>(nf, wb_e1, enc_b1, t0, N);
  gemm_bf16_nh<4, 0><<<gN, 256, 0, stream>>>(t0, wb_e2, enc_b2, t1, N);

  const long long totNH = (long long)N * HDIM;
  const int gNH  = (int)((totNH + 255) / 256);
  const int gAgg = (int)(((long long)(E + N) * 32 + 255) / 256);

  // three GCN layers
  float* xin = t1;
  float* xout = t2;
  for (int l = 0; l < 3; ++l) {
    gemm_bf16_nh<4, 0><<<gN, 256, 0, stream>>>(xin, wb_c[l], nullptr, t0, N);
    k_init_bias<<<gNH, 256, 0, stream>>>(xout, conv_b[l], totNH);
    k_aggregate<<<gAgg, 256, 0, stream>>>(t0, srci, dsti, dis, xout, E, N);
    if (l < 2) k_relu<<<gNH, 256, 0, stream>>>(xout, totNH);
    float* tmp = xin; xin = xout; xout = tmp;
  }

  // fused edge classifier
  const int gE = (E + 255) / 256;   // 256 edges per block (RS=2)
  edge_cls<<<gE, 256, 0, stream>>>(xin, srci, dsti, wb_cls, cls_b1, cls_w2, cls_b2, out, E);
}
